// MinGRU_39719857553536
// MI455X (gfx1250) — matrix operand order repaired and checked by reading, not compiled, not run on hardware
//
#include <hip/hip_runtime.h>
#include <hip/hip_bf16.h>

typedef __bf16 bf16_t;
typedef bf16_t v16bf  __attribute__((ext_vector_type(16)));
typedef bf16_t bf16x8 __attribute__((ext_vector_type(8)));
typedef float  v8f    __attribute__((ext_vector_type(8)));
typedef unsigned int u32x4 __attribute__((ext_vector_type(4)));
typedef int          i32x8 __attribute__((ext_vector_type(8)));
typedef int          i32x4 __attribute__((ext_vector_type(4)));

// Problem sizes from the reference: B=4, S=4096, D=1024, H=1024
constexpr int B_  = 4;
constexpr int S_  = 4096;
constexpr int D_  = 1024;
constexpr int H_  = 1024;
constexpr int M_  = B_ * S_;        // 16384 rows of x
// GEMM tiling: block tile 128x32, 8 waves, each wave 32(M)x16(N) for both GEMMs
constexpr int BM = 128, BN = 32, BK = 32;
constexpr int KSTEPS = D_ / BK;     // 32
constexpr int LDA = BK + 8;         // LDS row pitch (bf16) = 80B, matches TDM pad
// Scan chunking: S = CCH chunks of CT steps
constexpr int CCH = 64;
constexpr int CT  = S_ / CCH;       // 64

#if defined(__gfx1250__) && __has_builtin(__builtin_amdgcn_tensor_load_to_lds)
#define USE_TDM 1
#else
#define USE_TDM 0
#endif

__device__ __forceinline__ float softplus_f(float x) {
  return fmaxf(x, 0.f) + log1pf(__expf(-fabsf(x)));
}
__device__ __forceinline__ float logg_f(float x) {
  return (x >= 0.f) ? __logf(x + 0.5f) : -softplus_f(-x);
}
__device__ __forceinline__ float logaddexp_f(float a, float b) {
  float m = fmaxf(a, b);
  if (m == -__builtin_inff()) return -__builtin_inff();
  return m + log1pf(__expf(fminf(a, b) - m));
}

#if USE_TDM
// ---------------------------------------------------------------------------
// Issue one TDM 2D tile load (global bf16 -> LDS), with LDS padding of 16B
// after every 64B row so rows land at pitch 80B (= LDA bf16 elems).
// Descriptor packing per CDNA5 ISA.
// Toolchain here ships the 6-arg builtin:
//   (u32x4 g0, i32x8 g1, i32x4, i32x4, i32x8, i32 cpol)
// Groups beyond 0/1 are only needed for >2D tensors (tile_dim2==0 -> unused),
// so they are passed as zeros.
// ---------------------------------------------------------------------------
__device__ __forceinline__ void tdm_load_tile(unsigned lds_addr,
                                              const bf16_t* gptr,
                                              int tile_w, int tile_h,
                                              int row_stride_elems) {
  const unsigned long long ga = (unsigned long long)(uintptr_t)gptr;
  u32x4 g0;
  g0[0] = 1u;                                    // count=1 (valid), user mode
  g0[1] = lds_addr;                              // LDS byte address
  g0[2] = (unsigned)(ga & 0xffffffffu);          // global_addr[31:0]
  g0[3] = (unsigned)((ga >> 32) & 0x01ffffffu)   // global_addr[56:32]
          | (2u << 30);                          // type = 2 ("image")
  i32x8 g1;
  const unsigned dim0 = (unsigned)row_stride_elems;  // tensor_dim0 (elems)
  const unsigned dim1 = 1u << 24;                    // tensor_dim1: large
  g1[0] = (int)((1u << 16)      // data_size = 1 -> 2 bytes
              | (1u << 20)      // pad_enable
              | (3u << 22)      // pad_interval code 3 -> every 16 DWORDs (64B)
              | (3u << 25));    // pad_amount  code 3 -> 4 DWORDs (16B)
  g1[1] = (int)((dim0 & 0xffffu) << 16);                      // dim0[15:0]
  g1[2] = (int)((dim0 >> 16) | ((dim1 & 0xffffu) << 16));     // dim0[31:16]|dim1[15:0]
  g1[3] = (int)((dim1 >> 16) | ((unsigned)tile_w << 16));     // dim1[31:16]|tile_dim0
  g1[4] = (int)((unsigned)tile_h & 0xffffu);                  // tile_dim1 (tile_dim2=0)
  g1[5] = (int)(unsigned)row_stride_elems;                    // dim0_stride[31:0]
  g1[6] = 0;
  g1[7] = 0;
  const i32x4 gz4 = {0, 0, 0, 0};
  const i32x8 gz8 = {0, 0, 0, 0, 0, 0, 0, 0};
  __builtin_amdgcn_tensor_load_to_lds(g0, g1, gz4, gz4, gz8, 0);
}
#endif

// ---------------------------------------------------------------------------
// Pre-pass: f32 -> bf16 (RNE) conversion so the GEMM can stream bf16 via TDM.
// ---------------------------------------------------------------------------
__global__ __launch_bounds__(256)
void cvt_bf16_kernel(const float* __restrict__ in, bf16_t* __restrict__ out,
                     int n8) {
  const int i = blockIdx.x * 256 + threadIdx.x;
  if (i >= n8) return;
  const float4 a = reinterpret_cast<const float4*>(in)[i * 2 + 0];
  const float4 b = reinterpret_cast<const float4*>(in)[i * 2 + 1];
  bf16x8 o;
  o[0] = (bf16_t)a.x; o[1] = (bf16_t)a.y; o[2] = (bf16_t)a.z; o[3] = (bf16_t)a.w;
  o[4] = (bf16_t)b.x; o[5] = (bf16_t)b.y; o[6] = (bf16_t)b.z; o[7] = (bf16_t)b.w;
  reinterpret_cast<bf16x8*>(out)[i] = o;
}

// ---------------------------------------------------------------------------
// Kernel 1: fused dual-GEMM (bf16 WMMA, f32 accumulate) + log-space epilogue.
// Block 256 threads = 8 waves (4 in M x 2 in N); wave tile 32x16, 4 WMMA/k-step.
// Tiles staged into LDS by the Tensor Data Mover, double-buffered on TENSORcnt.
// ---------------------------------------------------------------------------
__global__ __launch_bounds__(256)
void mingru_gemm_kernel(const bf16_t* __restrict__ xb,
                        const bf16_t* __restrict__ wzb,
                        const bf16_t* __restrict__ whb,
                        const float* __restrict__ bz,
                        const float* __restrict__ bh,
                        float* __restrict__ LC,
                        float* __restrict__ LV) {
  __shared__ bf16_t sX [2][BM][LDA];
  __shared__ bf16_t sWz[2][BN][LDA];
  __shared__ bf16_t sWh[2][BN][LDA];

  const int t       = threadIdx.x;
  const int blockN0 = blockIdx.x * BN;
  const int blockM0 = blockIdx.y * BM;

  const int wave  = t >> 5;
  const int lane  = t & 31;
  const int half  = lane >> 4;         // 0: lanes 0-15, 1: lanes 16-31
  const int l16   = lane & 15;
  const int waveM = (wave & 3) * 32;   // 0,32,64,96
  const int waveN = (wave >> 2) * 16;  // 0,16

  v8f accZ0 = {}, accZ1 = {}, accH0 = {}, accH1 = {};

#if USE_TDM
  auto issueStage = [&](int ks, int buf) {
    const int kk = ks * BK;
    tdm_load_tile((unsigned)(uintptr_t)&sX[buf][0][0],
                  xb + (size_t)blockM0 * D_ + kk, BK, BM, D_);
    tdm_load_tile((unsigned)(uintptr_t)&sWz[buf][0][0],
                  wzb + (size_t)blockN0 * D_ + kk, BK, BN, D_);
    tdm_load_tile((unsigned)(uintptr_t)&sWh[buf][0][0],
                  whb + (size_t)blockN0 * D_ + kk, BK, BN, D_);
  };
  if (t < 32) issueStage(0, 0);   // wave 0 drives the TDM (issued once per wave)
#else
  auto issueStage = [&](int ks, int buf) {
    const int kk = ks * BK;
    // cooperative bf16x8 copies (fallback only)
    for (int i = t; i < BM * (BK / 8); i += 256) {
      const int row = i / (BK / 8), c8 = (i % (BK / 8)) * 8;
      *reinterpret_cast<bf16x8*>(&sX[buf][row][c8]) =
          *reinterpret_cast<const bf16x8*>(&xb[(size_t)(blockM0 + row) * D_ + kk + c8]);
    }
    for (int i = t; i < BN * (BK / 8); i += 256) {
      const int row = i / (BK / 8), c8 = (i % (BK / 8)) * 8;
      *reinterpret_cast<bf16x8*>(&sWz[buf][row][c8]) =
          *reinterpret_cast<const bf16x8*>(&wzb[(size_t)(blockN0 + row) * D_ + kk + c8]);
      *reinterpret_cast<bf16x8*>(&sWh[buf][row][c8]) =
          *reinterpret_cast<const bf16x8*>(&whb[(size_t)(blockN0 + row) * D_ + kk + c8]);
    }
  };
  issueStage(0, 0);
  __syncthreads();
#endif

  for (int ks = 0; ks < KSTEPS; ++ks) {
    const int cur = ks & 1;
#if USE_TDM
    if (t < 32) {
      if (ks + 1 < KSTEPS) {
        issueStage(ks + 1, cur ^ 1);              // 3 more in flight (total 6)
        __builtin_amdgcn_s_wait_tensorcnt(3);     // stage ks complete (in-order)
      } else {
        __builtin_amdgcn_s_wait_tensorcnt(0);
      }
    }
    __syncthreads();                              // release waves 1..7
#else
    if (ks + 1 < KSTEPS) issueStage(ks + 1, cur ^ 1);
#endif

    // A fragments (16x32 bf16): lane<16 -> row l16, K {0..7,16..23};
    //                           lane>=16 -> row l16, K {8..15,24..31}
    const int aRow = waveM + l16;
    const int ak0  = half ? 8 : 0;
    const bf16x8 a0l = *reinterpret_cast<const bf16x8*>(&sX[cur][aRow][ak0]);
    const bf16x8 a0h = *reinterpret_cast<const bf16x8*>(&sX[cur][aRow][ak0 + 16]);
    const bf16x8 a1l = *reinterpret_cast<const bf16x8*>(&sX[cur][aRow + 16][ak0]);
    const bf16x8 a1h = *reinterpret_cast<const bf16x8*>(&sX[cur][aRow + 16][ak0 + 16]);
    v16bf a0, a1;
#pragma unroll
    for (int i = 0; i < 8; ++i) {
      a0[i] = a0l[i]; a0[i + 8] = a0h[i];
      a1[i] = a1l[i]; a1[i + 8] = a1h[i];
    }

    // B fragments (32x16 bf16): lane holds column N=l16,
    //   K {0..7,16..23} (lanes 0-15) or {8..15,24..31} (lanes 16-31); W stored [n][k].
    const int bRow = waveN + l16;
    const int bk0  = half ? 8 : 0;
    const bf16x8 bzl = *reinterpret_cast<const bf16x8*>(&sWz[cur][bRow][bk0]);
    const bf16x8 bzh = *reinterpret_cast<const bf16x8*>(&sWz[cur][bRow][bk0 + 16]);
    const bf16x8 bhl = *reinterpret_cast<const bf16x8*>(&sWh[cur][bRow][bk0]);
    const bf16x8 bhh = *reinterpret_cast<const bf16x8*>(&sWh[cur][bRow][bk0 + 16]);
    v16bf bzf, bhf;
#pragma unroll
    for (int i = 0; i < 8; ++i) {
      bzf[i] = bzl[i]; bzf[i + 8] = bzh[i];
      bhf[i] = bhl[i]; bhf[i + 8] = bhh[i];
    }

    accZ0 = __builtin_amdgcn_wmma_f32_16x16x32_bf16(
        false, a0, false, bzf, (short)0, accZ0, false, false);
    accH0 = __builtin_amdgcn_wmma_f32_16x16x32_bf16(
        false, a0, false, bhf, (short)0, accH0, false, false);
    accZ1 = __builtin_amdgcn_wmma_f32_16x16x32_bf16(
        false, a1, false, bzf, (short)0, accZ1, false, false);
    accH1 = __builtin_amdgcn_wmma_f32_16x16x32_bf16(
        false, a1, false, bhf, (short)0, accH1, false, false);

    __syncthreads();
  }

  // Epilogue: C/D layout -> lane gives N=l16(+waveN), element e gives M=half*8+e.
  const int   nG  = blockN0 + waveN + l16;
  const float bzv = bz[nG];
  const float bhv = bh[nG];
#pragma unroll
  for (int sub = 0; sub < 2; ++sub) {
    const v8f az = sub ? accZ1 : accZ0;
    const v8f ah = sub ? accH1 : accH0;
#pragma unroll
    for (int e = 0; e < 8; ++e) {
      const int mG = blockM0 + waveM + sub * 16 + half * 8 + e;
      const float k  = az[e] + bzv;
      const float th = ah[e] + bhv;
      const float lc = -softplus_f(k);
      const float lv = -softplus_f(-k) + logg_f(th);
      const size_t o = (size_t)mG * H_ + nG;
      LC[o] = lc;
      LV[o] = lv;
    }
  }
}

// ---------------------------------------------------------------------------
// Scan pass 1: per-chunk carry (A = sum lc, B = chunk-local logaddexp scan end)
// ---------------------------------------------------------------------------
__global__ __launch_bounds__(256)
void mingru_chunk1(const float* __restrict__ LC, const float* __restrict__ LV,
                   float* __restrict__ cA, float* __restrict__ cB) {
  const int tid = blockIdx.x * 256 + threadIdx.x;
  const int h = tid % H_;
  const int c = (tid / H_) % CCH;
  const int b = tid / (H_ * CCH);
  const size_t base = ((size_t)b * S_ + (size_t)c * CT) * H_ + h;
  float A  = 0.f;
  float Bl = -__builtin_inff();
  for (int tt = 0; tt < CT; ++tt) {
    const float lc = LC[base + (size_t)tt * H_];
    const float lv = LV[base + (size_t)tt * H_];
    Bl = logaddexp_f(lc + Bl, lv);
    A += lc;
  }
  const size_t o = ((size_t)b * CCH + c) * H_ + h;
  cA[o] = A;
  cB[o] = Bl;
}

// ---------------------------------------------------------------------------
// Scan pass 2: sequential combine across chunks (64 deep); exclusive states.
// ---------------------------------------------------------------------------
__global__ __launch_bounds__(256)
void mingru_chunk2(const float* __restrict__ h0,
                   const float* __restrict__ cA, const float* __restrict__ cB,
                   float* __restrict__ st0) {
  const int tid = blockIdx.x * 256 + threadIdx.x;
  const int h = tid % H_;
  const int b = tid / H_;
  float st = logg_f(h0[(size_t)b * H_ + h]);
  for (int c = 0; c < CCH; ++c) {
    const size_t o = ((size_t)b * CCH + c) * H_ + h;
    st0[o] = st;
    st = logaddexp_f(cA[o] + st, cB[o]);
  }
}

// ---------------------------------------------------------------------------
// Scan pass 3: replay chunk with true initial state, write exp(log_h) outputs.
// ---------------------------------------------------------------------------
__global__ __launch_bounds__(256)
void mingru_chunk3(const float* __restrict__ LC, const float* __restrict__ LV,
                   const float* __restrict__ st0,
                   float* __restrict__ out, float* __restrict__ outLast) {
  const int tid = blockIdx.x * 256 + threadIdx.x;
  const int h = tid % H_;
  const int c = (tid / H_) % CCH;
  const int b = tid / (H_ * CCH);
  float lh = st0[((size_t)b * CCH + c) * H_ + h];
  const size_t base = ((size_t)b * S_ + (size_t)c * CT) * H_ + h;
  for (int tt = 0; tt < CT; ++tt) {
    lh = logaddexp_f(LC[base + (size_t)tt * H_] + lh, LV[base + (size_t)tt * H_]);
    out[base + (size_t)tt * H_] = __expf(lh);
  }
  if (c == CCH - 1) outLast[(size_t)b * H_ + h] = __expf(lh);
}

extern "C" void kernel_launch(void* const* d_in, const int* in_sizes, int n_in,
                              void* d_out, int out_size, void* d_ws, size_t ws_size,
                              hipStream_t stream) {
  (void)in_sizes; (void)n_in; (void)out_size; (void)ws_size;
  const float* x  = (const float*)d_in[0];   // (B,S,D)
  const float* h0 = (const float*)d_in[1];   // (B,1,H)
  const float* Wz = (const float*)d_in[2];   // (H,D)
  const float* bz = (const float*)d_in[3];   // (H)
  const float* Wh = (const float*)d_in[4];   // (H,D)
  const float* bh = (const float*)d_in[5];   // (H)
  float* out = (float*)d_out;                // (B,S,H) ++ (B,1,H)

  // Workspace layout
  bf16_t* xb  = (bf16_t*)d_ws;                        // M_*D_ bf16 (32MB)
  bf16_t* wzb = xb  + (size_t)M_ * D_;                // H_*D_ bf16 (2MB)
  bf16_t* whb = wzb + (size_t)H_ * D_;                // H_*D_ bf16 (2MB)
  float*  LC  = (float*)(whb + (size_t)H_ * D_);      // B*S*H f32 (64MB)
  float*  LV  = LC  + (size_t)M_ * H_;                // B*S*H f32 (64MB)
  float*  cA  = LV  + (size_t)M_ * H_;                // B*CCH*H f32
  float*  cB  = cA  + (size_t)B_ * CCH * H_;
  float*  st0 = cB  + (size_t)B_ * CCH * H_;

  // f32 -> bf16 pre-pass
  const int nx8 = M_ * D_ / 8;
  const int nw8 = H_ * D_ / 8;
  cvt_bf16_kernel<<<nx8 / 256, 256, 0, stream>>>(x,  xb,  nx8);
  cvt_bf16_kernel<<<nw8 / 256, 256, 0, stream>>>(Wz, wzb, nw8);
  cvt_bf16_kernel<<<nw8 / 256, 256, 0, stream>>>(Wh, whb, nw8);

  dim3 g1(H_ / BN, M_ / BM);                          // 32 x 128 blocks
  mingru_gemm_kernel<<<g1, 256, 0, stream>>>(xb, wzb, whb, bz, bh, LC, LV);

  const int n1 = B_ * CCH * H_;                       // 262144 threads
  mingru_chunk1<<<n1 / 256, 256, 0, stream>>>(LC, LV, cA, cB);

  const int n2 = B_ * H_;                             // 4096 threads
  mingru_chunk2<<<n2 / 256, 256, 0, stream>>>(h0, cA, cB, st0);

  mingru_chunk3<<<n1 / 256, 256, 0, stream>>>(LC, LV, st0, out,
                                              out + (size_t)M_ * H_);
}
